// Segmentor_37271726194925
// MI455X (gfx1250) — matrix operand order repaired and checked by reading, not compiled, not run on hardware
//
#include <hip/hip_runtime.h>
#include <cstdint>

// ---------------------------------------------------------------------------
// Types / helpers
// ---------------------------------------------------------------------------
typedef _Float16 f16;
typedef __attribute__((ext_vector_type(16))) _Float16 v16h;
typedef __attribute__((ext_vector_type(8)))  float    v8f;

union F16Frag {
    v16h     v;
    f16      h[16];
    uint32_t u[8];
    uint4    q[2];
};

__device__ __forceinline__ float preluf(float x, float a) { return x >= 0.f ? x : a * x; }
__device__ __forceinline__ float sigm(float x) { return 1.f / (1.f + __expf(-x)); }

__device__ __forceinline__ v8f wmma_f16(const F16Frag& a, const F16Frag& b, v8f c) {
    // D = A(16x32 f16) * B(32x16 f16) + C(16x16 f32)
    return __builtin_amdgcn_wmma_f32_16x16x32_f16(false, a.v, false, b.v,
                                                  (short)0, c, false, false);
}

// ---------------------------------------------------------------------------
// Kernel: swizzle an (N x K) row-major weight (W[n*ldW + colOff + k]) into
// WMMA B-fragment layout: frag[((kt*nNt + nt)*32 + lane)*8 + p] packs the
// f16 pair  B[kt*32 + 2*(p&3) + 16*(p>>2) + 8*hi][nt*16 + lo]  (K-major pairs per dword).
// ---------------------------------------------------------------------------
__global__ void k_swiz_b(const float* __restrict__ W, uint32_t* __restrict__ frag,
                         int Nvalid, int ldW, int colOff, int nKt, int nNt)
{
    int idx = blockIdx.x * blockDim.x + threadIdx.x;
    int total = nKt * nNt * 32 * 8;
    if (idx >= total) return;
    int p    = idx & 7;
    int lane = (idx >> 3) & 31;
    int t    = idx >> 8;            // kt*nNt + nt
    int nt   = t % nNt;
    int kt   = t / nNt;
    int lo = lane & 15, hi = lane >> 4;
    int n  = nt * 16 + lo;
    int k0 = kt * 32 + (p & 3) * 2 + ((p >> 2) * 16) + hi * 8;
    F16Frag fr;
    fr.h[0] = (f16)0.f;
    fr.h[1] = (f16)0.f;
    if (n < Nvalid) {
        fr.h[0] = (f16)W[n * ldW + colOff + k0];
        fr.h[1] = (f16)W[n * ldW + colOff + k0 + 1];
    }
    frag[idx] = fr.u[0];
}

// ---------------------------------------------------------------------------
// Kernel: input-to-hidden pre-activations  pre[(s*2+b)*512 + g] =
//         b[g] + sum_k X[(b*256+s)*K + k] * Wih[g*K + k]
// ---------------------------------------------------------------------------
__global__ void k_pre(const float* __restrict__ X, const float* __restrict__ Wih,
                      const float* __restrict__ bias, float* __restrict__ pre, int K)
{
    int idx = blockIdx.x * blockDim.x + threadIdx.x;
    int g  = idx & 511;
    int sb = idx >> 9;
    int b  = sb & 1;
    int s  = sb >> 1;
    if (s >= 256) return;
    const float* x = X + (b * 256 + s) * K;
    const float* w = Wih + g * K;
    float acc = bias[g];
    for (int k = 0; k < K; ++k) acc += x[k] * w[k];
    pre[(s * 2 + b) * 512 + g] = acc;
}

// ---------------------------------------------------------------------------
// Kernel: LSTM recurrence, one workgroup per direction (blockIdx.x = dir).
// 1024 threads = 32 waves; wave w owns n-tile w of the (M=16, K=128, N=512)
// recurrent GEMM  g = h @ Whh^T, computed with v_wmma per step. Each wave
// keeps only 4 B-fragments (32 VGPRs) resident -> no scratch spills.
// ---------------------------------------------------------------------------
__global__ void __launch_bounds__(1024, 1)
k_lstm(const float* __restrict__ preF, const float* __restrict__ preB,
       const uint32_t* __restrict__ fragBase, float* __restrict__ out)
{
    int dir = blockIdx.x;                       // 0 = fwd, 1 = bwd
    const float*    pre  = dir ? preB : preF;
    const uint32_t* frag = fragBase + dir * 32768;

    __shared__ f16   hF[16][128];               // h padded to 16 rows (rows>=2 stay 0)
    __shared__ float cS[2][128];
    __shared__ float gS[2][512];

    int tid  = threadIdx.x;
    int lane = tid & 31, wave = tid >> 5;       // wave = n-tile (0..31)
    int lo   = lane & 15, hi = lane >> 4;

    for (int t = tid; t < 16 * 128; t += 1024) ((f16*)hF)[t] = (f16)0.f;
    if (tid < 256) cS[tid >> 7][tid & 127] = 0.f;
    __syncthreads();

    // Load resident Whh fragments for this wave's n-tile: [kstep(4)]
    F16Frag bf[4];
#pragma unroll
    for (int ks = 0; ks < 4; ++ks) {
        const uint32_t* fp = frag + ((ks * 32 + wave) * 32 + lane) * 8;
        bf[ks].q[0] = *(const uint4*)fp;
        bf[ks].q[1] = *(const uint4*)(fp + 4);
    }

    for (int t = 0; t < 256; ++t) {
        int s = dir ? (255 - t) : t;

        // Build A fragments from h (f16 pairs read as aligned LDS dwords)
        F16Frag af[4];
#pragma unroll
        for (int ks = 0; ks < 4; ++ks)
#pragma unroll
            for (int p = 0; p < 8; ++p) {
                int k0 = ks * 32 + (p & 3) * 2 + ((p >> 2) * 16) + hi * 8;
                af[ks].u[p] = *(const uint32_t*)&hF[lo][k0];
            }

        v8f acc = {};
#pragma unroll
        for (int ks = 0; ks < 4; ++ks) acc = wmma_f16(af[ks], bf[ks], acc);
        if (hi == 0) {                          // rows M=0 (acc[0]) and M=1 (acc[1])
            int n = wave * 16 + lo;
            gS[0][n] = acc[0];
            gS[1][n] = acc[1];
        }
        __syncthreads();

        if (tid < 256) {                        // elementwise gates: thread -> (b, j)
            int b = tid >> 7, j = tid & 127;
            const float* pr = pre + (s * 2 + b) * 512;
            float gi = gS[b][j]       + pr[j];
            float gf = gS[b][128 + j] + pr[128 + j];
            float gg = gS[b][256 + j] + pr[256 + j];
            float go = gS[b][384 + j] + pr[384 + j];
            float c = sigm(gf) * cS[b][j] + sigm(gi) * tanhf(gg);
            float h = sigm(go) * tanhf(c);
            cS[b][j] = c;
            hF[b][j] = (f16)h;
            out[(b * 256 + s) * 256 + dir * 128 + j] = h;
        }
        __syncthreads();
    }
}

// ---------------------------------------------------------------------------
// Kernel: cumsum over time of rnn_out -> cum
// ---------------------------------------------------------------------------
__global__ void k_cumsum(const float* __restrict__ rnn, float* __restrict__ cum)
{
    int idx = blockIdx.x * blockDim.x + threadIdx.x;   // b*256 + k
    if (idx >= 512) return;
    int b = idx >> 8, k = idx & 255;
    float s = 0.f;
    for (int t = 0; t < 256; ++t) {
        s += rnn[(b * 256 + t) * 256 + k];
        cum[(b * 256 + t) * 256 + k] = s;
    }
}

// ---------------------------------------------------------------------------
// Kernel: separable score terms Pd (with b_s1 folded) and Pe, padded to N=112
// ---------------------------------------------------------------------------
__global__ void k_pdpe(const float* __restrict__ rnn, const float* __restrict__ w_s1,
                       const float* __restrict__ b_s1, const float* __restrict__ a_s0p,
                       float* __restrict__ Pd, float* __restrict__ Pe)
{
    int idx = blockIdx.x * blockDim.x + threadIdx.x;   // which*512*112 + bs*112 + n
    int n     = idx % 112;
    int bs    = (idx / 112) % 512;
    int which = idx / (112 * 512);
    if (which >= 2) return;
    float a0 = a_s0p[0];
    float acc = 0.f;
    if (n < 100) {
        const float* w = w_s1 + n * 768 + (which ? 512 : 256);
        const float* x = rnn + bs * 256;
        for (int k = 0; k < 256; ++k) acc += w[k] * preluf(x[k], a0);
        if (!which) acc += b_s1[n];
    }
    (which ? Pe : Pd)[bs * 112 + n] = acc;
}

// ---------------------------------------------------------------------------
// Kernel: pairwise scores. One workgroup per (b, i) = blockIdx.x.
// Wave w handles j-tiles {2w, 2w+1}; each tile is a WMMA GEMM
// [M=16 j-rows] x [N=112] x [K=256] on A = prelu(cum[j]-cum[i]) in f16.
// Epilogue adds Pd+Pe, applies prelu, dots with w_s2 via LDS f32 atomics.
// ---------------------------------------------------------------------------
__global__ void __launch_bounds__(256)
k_scores(const float* __restrict__ cum, const uint32_t* __restrict__ w1c_frag,
         const float* __restrict__ Pd, const float* __restrict__ Pe,
         const float* __restrict__ w_s2, const float* __restrict__ b_s2,
         const float* __restrict__ a_s0p, const float* __restrict__ a_s1p,
         float* __restrict__ sc)
{
    int bi = blockIdx.x;
    int b = bi >> 8, i = bi & 255;
    int tid = threadIdx.x, lane = tid & 31, wave = tid >> 5;
    int lo = lane & 15, hi = lane >> 4;

    __shared__ float cumI[256];
    __shared__ float ws2[112];
    __shared__ float part[256];

    float a0 = a_s0p[0], a1 = a_s1p[0];
    cumI[tid] = cum[(b * 256 + i) * 256 + tid];
    if (tid < 112) ws2[tid] = (tid < 100) ? w_s2[tid] : 0.f;
    part[tid] = 0.f;

    // Prefetch this lane's A-operand rows for both tiles (gfx1250 global_prefetch)
    {
        int j0 = (wave * 2) * 16 + lo;
        __builtin_prefetch(cum + (b * 256 + j0) * 256, 0, 1);
        __builtin_prefetch(cum + (b * 256 + j0 + 16) * 256, 0, 1);
    }
    __syncthreads();

    const float* PdI = Pd + (b * 256 + i) * 112;

    for (int tile = 0; tile < 2; ++tile) {
        int jbase = (wave * 2 + tile) * 16;
        v8f acc[7] = {};
        int jA = jbase + lo;
        const float* cumJ = cum + (b * 256 + jA) * 256;

#pragma unroll
        for (int ks = 0; ks < 8; ++ks) {
            F16Frag af;
#pragma unroll
            for (int p = 0; p < 8; ++p) {
                int k0 = ks * 32 + (p & 3) * 2 + ((p >> 2) * 16) + hi * 8;
                float2 cj = *(const float2*)(cumJ + k0);
                af.h[2 * p]     = (f16)preluf(cj.x - cumI[k0],     a0);
                af.h[2 * p + 1] = (f16)preluf(cj.y - cumI[k0 + 1], a0);
            }
#pragma unroll
            for (int nt = 0; nt < 7; ++nt) {
                F16Frag bfr;
                const uint32_t* fp = w1c_frag + ((ks * 7 + nt) * 32 + lane) * 8;
                bfr.q[0] = *(const uint4*)fp;
                bfr.q[1] = *(const uint4*)(fp + 4);
                acc[nt] = wmma_f16(af, bfr, acc[nt]);
            }
        }

        // Epilogue: h = acc + Pd[i] + Pe[j]; score += w_s2 * prelu(h)
#pragma unroll
        for (int nt = 0; nt < 7; ++nt) {
            int n = nt * 16 + lo;
            float pd = PdI[n];
            float w2 = ws2[n];
#pragma unroll
            for (int r = 0; r < 8; ++r) {
                int j = jbase + hi * 8 + r;                 // C layout: M = r + 8*hi
                float h = acc[nt][r] + pd + Pe[(b * 256 + j) * 112 + n];
                atomicAdd(&part[j], w2 * preluf(h, a1));
            }
        }
    }
    __syncthreads();
    sc[(b * 256 + i) * 256 + tid] = part[tid] + b_s2[0];
}

// ---------------------------------------------------------------------------
// Kernel: DP over segments (sequential in i; single workgroup, 128 threads)
// ---------------------------------------------------------------------------
__global__ void k_dp(const float* __restrict__ sc, int* __restrict__ backptr)
{
    __shared__ float best[2][256];
    __shared__ float rv[128];
    __shared__ int   ri[128];
    int tid = threadIdx.x;                 // 128
    int b = tid >> 6, j0 = tid & 63;
    for (int t = tid; t < 512; t += 128) best[t >> 8][t & 255] = 0.f;
    if (tid < 2) backptr[tid * 256] = 0;
    __syncthreads();

    for (int i = 1; i < 256; ++i) {
        int start = (i - 64 > 0) ? (i - 64) : 0;
        int j = start + j0;
        float cand = (j < i) ? best[b][j] + sc[(b * 256 + j) * 256 + i] : -1e9f;
        rv[tid] = cand;
        ri[tid] = j;
        __syncthreads();
        for (int off = 32; off >= 1; off >>= 1) {
            if (j0 < off && rv[tid + off] > rv[tid]) {
                rv[tid] = rv[tid + off];
                ri[tid] = ri[tid + off];
            }
            __syncthreads();
        }
        if (j0 == 0) {
            best[b][i] = rv[tid];
            backptr[b * 256 + i] = ri[tid];
        }
        __syncthreads();
    }
}

// ---------------------------------------------------------------------------
// Kernel: backtrack + bmask + path score
// ---------------------------------------------------------------------------
__global__ void k_backtrack(const float* __restrict__ sc, const int* __restrict__ backptr,
                            const int* __restrict__ lengths,
                            int* __restrict__ bmask, float* __restrict__ pred)
{
    int tid = threadIdx.x;                 // 512
    if (tid < 512) bmask[tid] = 0;
    __syncthreads();
    if (tid < 2) {
        int b = tid;
        int cur = lengths[b] - 1;
        float acc = 0.f;
        for (int t = 0; t < 256; ++t) {
            bmask[b * 256 + cur] = 1;
            int prev = backptr[b * 256 + cur];
            bool active = cur > 0;
            if (active) acc += sc[(b * 256 + prev) * 256 + cur];
            cur = active ? prev : 0;
        }
        pred[b] = acc;
    }
}

// ---------------------------------------------------------------------------
// Kernel: classification + binary heads (per (b,s) workgroup of 192 threads)
// ---------------------------------------------------------------------------
__global__ void k_heads(const float* __restrict__ rnn,
                        const float* a_c0, const float* w_c1, const float* b_c1,
                        const float* a_c1, const float* w_c2, const float* b_c2,
                        const float* a_b0, const float* w_b1, const float* b_b1,
                        const float* a_b1, const float* w_b2, const float* b_b2,
                        float* __restrict__ out_cls, float* __restrict__ out_bin)
{
    int bs = blockIdx.x;
    int tid = threadIdx.x;                 // 192
    __shared__ float xr[256];
    __shared__ float hc[80], hb[80];
    for (int t = tid; t < 256; t += 192) xr[t] = rnn[bs * 256 + t];
    __syncthreads();
    if (tid < 80) {
        float a = a_c0[0], acc = b_c1[tid];
        const float* w = w_c1 + tid * 256;
        for (int k = 0; k < 256; ++k) acc += w[k] * preluf(xr[k], a);
        hc[tid] = acc;
    } else if (tid < 160) {
        int n = tid - 80;
        float a = a_b0[0], acc = b_b1[n];
        const float* w = w_b1 + n * 256;
        for (int k = 0; k < 256; ++k) acc += w[k] * preluf(xr[k], a);
        hb[n] = acc;
    }
    __syncthreads();
    if (tid < 40) {
        float a = a_c1[0], acc = b_c2[tid];
        const float* w = w_c2 + tid * 80;
        for (int k = 0; k < 80; ++k) acc += w[k] * preluf(hc[k], a);
        out_cls[bs * 40 + tid] = acc;
    } else if (tid < 42) {
        int m = tid - 40;
        float a = a_b1[0], acc = b_b2[m];
        const float* w = w_b2 + m * 80;
        for (int k = 0; k < 80; ++k) acc += w[k] * preluf(hb[k], a);
        out_bin[bs * 2 + m] = acc;
    }
}

// ---------------------------------------------------------------------------
// Host launcher
// ---------------------------------------------------------------------------
extern "C" void kernel_launch(void* const* d_in, const int* in_sizes, int n_in,
                              void* d_out, int out_size, void* d_ws, size_t ws_size,
                              hipStream_t stream)
{
    (void)in_sizes; (void)n_in; (void)out_size; (void)ws_size;

    const float* x        = (const float*)d_in[0];
    const int*   lengths  = (const int*)d_in[1];
    const float* w_ih[4]  = { (const float*)d_in[2], (const float*)d_in[5],
                              (const float*)d_in[8], (const float*)d_in[11] };
    const float* w_hh[4]  = { (const float*)d_in[3], (const float*)d_in[6],
                              (const float*)d_in[9], (const float*)d_in[12] };
    const float* b_l[4]   = { (const float*)d_in[4], (const float*)d_in[7],
                              (const float*)d_in[10], (const float*)d_in[13] };
    const float* a_s0 = (const float*)d_in[14];
    const float* w_s1 = (const float*)d_in[15];
    const float* b_s1 = (const float*)d_in[16];
    const float* a_s1 = (const float*)d_in[17];
    const float* w_s2 = (const float*)d_in[18];
    const float* b_s2 = (const float*)d_in[19];
    const float* a_c0 = (const float*)d_in[20];
    const float* w_c1 = (const float*)d_in[21];
    const float* b_c1 = (const float*)d_in[22];
    const float* a_c1 = (const float*)d_in[23];
    const float* w_c2 = (const float*)d_in[24];
    const float* b_c2 = (const float*)d_in[25];
    const float* a_b0 = (const float*)d_in[26];
    const float* w_b1 = (const float*)d_in[27];
    const float* b_b1 = (const float*)d_in[28];
    const float* a_b1 = (const float*)d_in[29];
    const float* w_b2 = (const float*)d_in[30];
    const float* b_b2 = (const float*)d_in[31];

    // Workspace layout (floats)
    float* ws = (float*)d_ws;
    uint32_t* whh_frag = (uint32_t*)ws;                 // 4 dirs * 32768 dwords
    uint32_t* w1c_frag = whh_frag + 4 * 32768;          // 14336 dwords
    float* pre_f  = (float*)(w1c_frag + 14336);         // 262144
    float* pre_b  = pre_f + 262144;                     // 262144
    float* rnn0   = pre_b + 262144;                     // 131072
    float* rnn1   = rnn0 + 131072;                      // 131072
    float* cum    = rnn1 + 131072;                      // 131072
    float* Pd     = cum + 131072;                       // 57344
    float* Pe     = Pd + 57344;                         // 57344
    float* sc     = Pe + 57344;                         // 131072
    int*   bptr   = (int*)(sc + 131072);                // 512

    // Output regions
    float* out      = (float*)d_out;
    float* out_cls  = out;                              // 20480
    float* out_bin  = out + 20480;                      // 1024
    int*   out_mask = (int*)(out + 21504);              // 512 int32
    float* out_pred = out + 22016;                      // 2

    // 1) Pre-swizzle WMMA B-fragments: Whh (4 dirs) and W1c block of w_s1
    for (int d = 0; d < 4; ++d)
        k_swiz_b<<<128, 256, 0, stream>>>(w_hh[d], whh_frag + d * 32768,
                                          512, 128, 0, 4, 32);
    k_swiz_b<<<56, 256, 0, stream>>>(w_s1, w1c_frag, 100, 768, 0, 8, 7);

    // 2) Layer 0 biLSTM
    k_pre<<<1024, 256, 0, stream>>>(x, w_ih[0], b_l[0], pre_f, 80);
    k_pre<<<1024, 256, 0, stream>>>(x, w_ih[1], b_l[1], pre_b, 80);
    k_lstm<<<2, 1024, 0, stream>>>(pre_f, pre_b, whh_frag, rnn0);

    // 3) Layer 1 biLSTM
    k_pre<<<1024, 256, 0, stream>>>(rnn0, w_ih[2], b_l[2], pre_f, 256);
    k_pre<<<1024, 256, 0, stream>>>(rnn0, w_ih[3], b_l[3], pre_b, 256);
    k_lstm<<<2, 1024, 0, stream>>>(pre_f, pre_b, whh_frag + 2 * 32768, rnn1);

    // 4) Heads
    k_heads<<<512, 192, 0, stream>>>(rnn1,
                                     a_c0, w_c1, b_c1, a_c1, w_c2, b_c2,
                                     a_b0, w_b1, b_b1, a_b1, w_b2, b_b2,
                                     out_cls, out_bin);

    // 5) Pairwise scores: cumsum, separable terms, WMMA GEMM
    k_cumsum<<<2, 256, 0, stream>>>(rnn1, cum);
    k_pdpe<<<448, 256, 0, stream>>>(rnn1, w_s1, b_s1, a_s0, Pd, Pe);
    k_scores<<<512, 256, 0, stream>>>(cum, w1c_frag, Pd, Pe, w_s2, b_s2,
                                      a_s0, a_s1, sc);

    // 6) DP + backtrack
    k_dp<<<1, 128, 0, stream>>>(sc, bptr);
    k_backtrack<<<1, 512, 0, stream>>>(sc, bptr, lengths, out_mask, out_pred);
}
